// Network_15607911154364
// MI455X (gfx1250) — matrix operand order repaired and checked by reading, not compiled, not run on hardware
//
// SepConv forward for MI455X (gfx1250, wave32, WMMA), NHWC f16 pipeline.
// All 3x3 convs = implicit GEMM on v_wmma_f32_16x16x32_f16 (f32 accumulate).
// K ordered (ky*3+kx, ci) over NHWC so every 16-wide K-run at a pixel is one
// aligned 32B global load; channel dims padded to multiples of 16 (join 6->16,
// subnet 51->64) so the vector path is used everywhere. LDS holds fragments in
// register order: fragment build = ds_load_b128 pairs, no shuffles.
// Block tile 64(Cout) x 128(pix) x 32(K); 8 waves; each wave = 2x2 WMMA tiles.
//
// Parameter leaf order: setup_inputs() dict insertion order, recursively:
// tensorFirst, tensorSecond, conv1..conv5, deconv5, up5, deconv4, up4,
// deconv3, up3, deconv2, up2, vert1, vert2, horiz1, horiz2; each conv = (w,b).

#include <hip/hip_runtime.h>
#include <hip/hip_bf16.h>

typedef _Float16 half_t;
typedef __attribute__((ext_vector_type(16))) _Float16 v16h;
typedef __attribute__((ext_vector_type(8)))  _Float16 v8h;
typedef __attribute__((ext_vector_type(8)))  float    v8f;

#define CONV_TM 64    // C_out tile per block
#define CONV_TN 128   // pixel tile per block
#define CONV_TK 32    // K chunk == WMMA K

// ---------------------------------------------------------------------------
// Implicit-GEMM 3x3 conv, stride 1, zero pad 1. Activations NHWC f16.
// Weights OIHW f32 (converted while staging). K index = g*Cin + ci, g=ky*3+kx.
//   Cin  : padded activation channel count (stride, multiple of 16)
//   CinW : real weight input channels (<= Cin; K-lanes past it contribute 0)
//   Cout : real outputs written;  CoutS : output channel stride
// ---------------------------------------------------------------------------
__global__ __launch_bounds__(256)
void conv3x3_wmma_kernel(const half_t* __restrict__ in,
                         const float*  __restrict__ wgt,
                         const float*  __restrict__ bias,
                         const half_t* __restrict__ skip,  // added after ReLU
                         half_t* __restrict__ out,
                         int B, int Cin, int CinW, int H, int W,
                         int Cout, int CoutS, int relu)
{
    // Fragments stored in register order: one lane's v16h is contiguous.
    __shared__ __attribute__((aligned(32))) half_t sA[4][32][16]; // 4 M-tiles
    __shared__ __attribute__((aligned(32))) half_t sB[8][32][16]; // 8 N-tiles

    const int hw     = H * W;
    const int Npix   = B * hw;
    const int Ktot   = Cin * 9;                 // Cin mult of 16 -> Ktot mult 16
    const int pBlock  = blockIdx.x * CONV_TN;
    const int coBlock = blockIdx.y * CONV_TM;

    const int tid  = threadIdx.x;
    const int lane = tid & 31;
    const int wave = tid >> 5;
    const int wrow = wave >> 2;                 // 0..1 -> M tiles 2*wrow+{0,1}
    const int wcol = wave & 3;                  // 0..3 -> N tiles 2*wcol+{0,1}

    // ---- per-thread staging coordinates (k-invariant parts hoisted) --------
    // B staging: one thread = one (ntile, lane) -> 16 consecutive K at a pixel
    const int sbNt = tid >> 5;
    const int sbL  = tid & 31;
    const int sbP  = pBlock + sbNt * 16 + (sbL & 15);
    const int sbKo = (sbL & 16);                // 0 or 16
    int sbB = 0, sbY = 0, sbX = 0;
    const bool sbPix = (sbP < Npix);
    if (sbPix) { sbB = sbP / hw; int r = sbP - sbB * hw; sbY = r / W; sbX = r - sbY * W; }

    // A staging: one thread = 8 consecutive K for one (mtile, lane, half)
    const int saMt = tid >> 6;                  // 0..3
    const int saL  = (tid >> 1) & 31;
    const int saJh = tid & 1;                   // j 0..7 or 8..15
    const int saCo = coBlock + saMt * 16 + (saL & 15);
    const int saKo = (saL & 16) + saJh * 8;

    v8f acc[2][2] = {};
    const size_t wtotal = (size_t)Cout * CinW * 9;

    for (int kc = 0; kc < Ktot; kc += CONV_TK) {
        if (kc + CONV_TK < Ktot) {              // L2 prefetch of next W chunk
            size_t q = (size_t)coBlock * CinW * 9 + (size_t)tid * 16;
            if (q < wtotal) __builtin_prefetch(&wgt[q], 0, 1);
        }

        __syncthreads();
        // ---- stage A (weights, f32 -> f16, fragment order) -----------------
        {
            int k0 = kc + saKo;
            v8h av = {};
            if (k0 < Ktot && saCo < Cout) {
                int g = k0 / Cin, ci = k0 - g * Cin;
#pragma unroll
                for (int jj = 0; jj < 8; ++jj) {
                    if (ci < CinW)
                        av[jj] = (half_t)wgt[((size_t)saCo * CinW + ci) * 9 + g];
                    ++ci;
                    if (ci == Cin) { ci = 0; ++g; }
                }
            }
            *(v8h*)&sA[saMt][saL][saJh * 8] = av;
        }
        // ---- stage B (NHWC input, one 32B vector load, fragment order) -----
        {
            int klo = kc + sbKo;
            v16h bv = {};
            if (sbPix && klo < Ktot) {
                int g0  = klo / Cin;             // whole run in one (ky,kx)
                int ci0 = klo - g0 * Cin;        // multiple of 16
                int gy  = g0 / 3;
                int iy  = sbY + gy - 1;
                int ix  = sbX + (g0 - gy * 3) - 1;
                if (iy >= 0 && iy < H && ix >= 0 && ix < W)
                    bv = *(const v16h*)(in +
                         ((size_t)(sbB * H + iy) * W + ix) * Cin + ci0);
            }
            *(v16h*)&sB[sbNt][sbL][0] = bv;
        }
        __syncthreads();

        // ---- fragments: 2x ds_load_b128 each; 4 WMMAs ----------------------
        v16h a0 = *(const v16h*)&sA[2 * wrow + 0][lane][0];
        v16h a1 = *(const v16h*)&sA[2 * wrow + 1][lane][0];
        v16h b0 = *(const v16h*)&sB[2 * wcol + 0][lane][0];
        v16h b1 = *(const v16h*)&sB[2 * wcol + 1][lane][0];
        acc[0][0] = __builtin_amdgcn_wmma_f32_16x16x32_f16(false, a0, false, b0,
                        (short)0, acc[0][0], false, false);
        acc[0][1] = __builtin_amdgcn_wmma_f32_16x16x32_f16(false, a0, false, b1,
                        (short)0, acc[0][1], false, false);
        acc[1][0] = __builtin_amdgcn_wmma_f32_16x16x32_f16(false, a1, false, b0,
                        (short)0, acc[1][0], false, false);
        acc[1][1] = __builtin_amdgcn_wmma_f32_16x16x32_f16(false, a1, false, b1,
                        (short)0, acc[1][1], false, false);
    }

    // ---- epilogue: NHWC, 8 consecutive Cout per lane -> one v8h store ------
    const bool fastC = ((CoutS & 7) == 0);
#pragma unroll
    for (int it = 0; it < 2; ++it) {
        int coBase = coBlock + (2 * wrow + it) * 16 + ((lane & 16) ? 8 : 0);
#pragma unroll
        for (int jt = 0; jt < 2; ++jt) {
            int p = pBlock + (2 * wcol + jt) * 16 + (lane & 15);
            if (p < Npix) {
                size_t ob = (size_t)p * CoutS;
                if (fastC && coBase + 8 <= Cout) {
                    v8h ov;
#pragma unroll
                    for (int r = 0; r < 8; ++r) {
                        float v = acc[it][jt][r] + bias[coBase + r];
                        if (relu) v = v > 0.f ? v : 0.f;
                        if (skip) v += (float)skip[ob + coBase + r];
                        ov[r] = (half_t)v;
                    }
                    *(v8h*)&out[ob + coBase] = ov;
                } else {
#pragma unroll
                    for (int r = 0; r < 8; ++r) {
                        int co = coBase + r;
                        if (co < Cout) {
                            float v = acc[it][jt][r] + bias[co];
                            if (relu) v = v > 0.f ? v : 0.f;
                            if (skip) v += (float)skip[ob + co];
                            out[ob + co] = (half_t)v;
                        }
                    }
                }
            }
        }
    }
}

// ---------------------------------------------------------------------------
// Elementwise helpers (NHWC f16 unless noted)
// ---------------------------------------------------------------------------
__global__ void fill0_kernel(half_t* p, long long n)
{
    long long i = (long long)blockIdx.x * blockDim.x + threadIdx.x;
    if (i < n) p[i] = (half_t)0.f;
}

// join: [B,384,384,16]; channels 0..2 <- t1, 3..5 <- t2, 6..15 <- 0
__global__ void pad_join_kernel(const float* __restrict__ t1,
                                const float* __restrict__ t2,
                                half_t* __restrict__ join,
                                int B, int Hin, int Win, int Hp, int Wp, int pad)
{
    long long idx = (long long)blockIdx.x * blockDim.x + threadIdx.x;
    long long total = (long long)B * Hp * Wp * 16;
    if (idx >= total) return;
    int c = (int)(idx & 15); long long t = idx >> 4;
    int x = (int)(t % Wp); t /= Wp;
    int y = (int)(t % Hp); int b = (int)(t / Hp);
    half_t v = (half_t)0.f;
    if (c < 6) {
        const float* src = (c < 3) ? t1 : t2;
        int sc = c % 3;
        int ys = y - pad; ys = ys < 0 ? 0 : (ys >= Hin ? Hin - 1 : ys);
        int xs = x - pad; xs = xs < 0 ? 0 : (xs >= Win ? Win - 1 : xs);
        v = (half_t)src[((size_t)(b * 3 + sc) * Hin + ys) * Win + xs];
    }
    join[idx] = v;
}

// Replication pad to [B,3,434,434] f32 (NCHW planes; best for sepconv rows)
__global__ void pad_rep_kernel(const float* __restrict__ t,
                               float* __restrict__ rep,
                               int B, int Hin, int Win, int Hr, int Wr, int pad)
{
    long long idx = (long long)blockIdx.x * blockDim.x + threadIdx.x;
    long long total = (long long)B * 3 * Hr * Wr;
    if (idx >= total) return;
    int x = (int)(idx % Wr); long long tt = idx / Wr;
    int y = (int)(tt % Hr); int bc = (int)(tt / Hr);
    int ys = y - pad; ys = ys < 0 ? 0 : (ys >= Hin ? Hin - 1 : ys);
    int xs = x - pad; xs = xs < 0 ? 0 : (xs >= Win ? Win - 1 : xs);
    rep[idx] = t[((size_t)bc * Hin + ys) * Win + xs];
}

__global__ void avgpool2_kernel(const half_t* __restrict__ in,
                                half_t* __restrict__ out, int B, int C, int H, int W)
{
    int Ho = H >> 1, Wo = W >> 1;
    long long idx = (long long)blockIdx.x * blockDim.x + threadIdx.x;
    long long total = (long long)B * Ho * Wo * C;
    if (idx >= total) return;
    int c = (int)(idx % C); long long t = idx / C;
    int x = (int)(t % Wo); t /= Wo;
    int y = (int)(t % Ho); int b = (int)(t / Ho);
    size_t i0 = ((size_t)(b * H + 2 * y) * W + 2 * x) * C + c;
    size_t rs = (size_t)W * C;
    float s = (float)in[i0] + (float)in[i0 + C] + (float)in[i0 + rs] + (float)in[i0 + rs + C];
    out[idx] = (half_t)(0.25f * s);
}

__global__ void upsample2_ac_kernel(const half_t* __restrict__ in,
                                    half_t* __restrict__ out, int B, int C, int H, int W)
{
    int Ho = 2 * H, Wo = 2 * W;
    long long idx = (long long)blockIdx.x * blockDim.x + threadIdx.x;
    long long total = (long long)B * Ho * Wo * C;
    if (idx >= total) return;
    int c = (int)(idx % C); long long t = idx / C;
    int xo = (int)(t % Wo); t /= Wo;
    int yo = (int)(t % Ho); int b = (int)(t / Ho);
    float py = (H > 1) ? yo * (float)(H - 1) / (float)(Ho - 1) : 0.f;
    float px = (W > 1) ? xo * (float)(W - 1) / (float)(Wo - 1) : 0.f;
    int y0 = (int)py, x0 = (int)px;
    int y1 = y0 + 1 < H ? y0 + 1 : H - 1;
    int x1 = x0 + 1 < W ? x0 + 1 : W - 1;
    float wy = py - y0, wx = px - x0;
    const half_t* base = in + (size_t)b * H * W * C;
    float v00 = (float)base[((size_t)y0 * W + x0) * C + c];
    float v01 = (float)base[((size_t)y0 * W + x1) * C + c];
    float v10 = (float)base[((size_t)y1 * W + x0) * C + c];
    float v11 = (float)base[((size_t)y1 * W + x1) * C + c];
    float v = (1.f - wy) * ((1.f - wx) * v00 + wx * v01)
            +        wy  * ((1.f - wx) * v10 + wx * v11);
    out[idx] = (half_t)v;
}

// ---------------------------------------------------------------------------
// Separable 51x51 application on the cropped region. Coeffs NHWC [B,384,384,51]
// (contiguous 51-channel reads). 16x16 output tile; 66x66 LDS input patch.
// ---------------------------------------------------------------------------
#define SEP_TS 16
#define SEP_K  51
__global__ __launch_bounds__(256)
void sepconv_kernel(const float*  __restrict__ rep,   // [B,3,434,434] f32
                    const half_t* __restrict__ vco,   // [B,384,384,51] f16
                    const half_t* __restrict__ hco,
                    float* __restrict__ out,          // [B,3,334,334] f32
                    int B, int Hout, int Wout, int Hr, int Wr, int Hc,
                    int accumulate)
{
    __shared__ float tile[SEP_TS + SEP_K - 1][SEP_TS + SEP_K - 1 + 2]; // 66x68
    const int bc  = blockIdx.z;
    const int b   = bc / 3;
    const int ty0 = blockIdx.y * SEP_TS;
    const int tx0 = blockIdx.x * SEP_TS;
    const int tid = threadIdx.y * SEP_TS + threadIdx.x;

    const float* src = rep + (size_t)bc * Hr * Wr;
    const int PS = SEP_TS + SEP_K - 1;
    for (int i = tid; i < PS * PS; i += 256) {
        int ry = i / PS, rx = i - ry * PS;
        int gy = ty0 + 25 + ry, gx = tx0 + 25 + rx;
        gy = gy < Hr ? gy : Hr - 1;
        gx = gx < Wr ? gx : Wr - 1;
        tile[ry][rx] = src[(size_t)gy * Wr + gx];
    }
    __syncthreads();

    const int yo = ty0 + threadIdx.y;
    const int xo = tx0 + threadIdx.x;
    if (yo >= Hout || xo >= Wout) return;

    const size_t cb = ((size_t)(b * Hc + (yo + 25)) * Hc + (xo + 25)) * SEP_K;

    float hv[SEP_K];
#pragma unroll
    for (int j = 0; j < SEP_K; ++j) hv[j] = (float)hco[cb + j];

    float acc = 0.f;
    for (int i = 0; i < SEP_K; ++i) {
        const float* row = &tile[threadIdx.y + i][threadIdx.x];
        float hi = 0.f;
#pragma unroll
        for (int j = 0; j < SEP_K; ++j) hi += row[j] * hv[j];
        acc += hi * (float)vco[cb + i];
    }

    size_t oi = ((size_t)bc * Hout + yo) * Wout + xo;
    if (accumulate) out[oi] += acc; else out[oi] = acc;
}

// ---------------------------------------------------------------------------
// Host orchestration
// ---------------------------------------------------------------------------
struct ConvP { const float* w; const float* b; };

static inline dim3 grid1d(long long total) { return dim3((unsigned)((total + 255) / 256)); }

static inline void conv3x3(hipStream_t s, const half_t* in, ConvP p,
                           const half_t* skip, half_t* out,
                           int B, int Cin, int CinW, int H, int W,
                           int Cout, int CoutS, int relu)
{
    int Npix = B * H * W;
    dim3 grid((Npix + CONV_TN - 1) / CONV_TN, (Cout + CONV_TM - 1) / CONV_TM, 1);
    conv3x3_wmma_kernel<<<grid, 256, 0, s>>>(in, p.w, p.b, skip, out,
                                             B, Cin, CinW, H, W, Cout, CoutS, relu);
}

static inline void basic3(hipStream_t s, const half_t* in, const ConvP* ps,
                          half_t* tA, half_t* tB, half_t* dst,
                          int B, int Cin, int H, int W, int Cout)
{
    conv3x3(s, in, ps[0], nullptr, tA,  B, Cin,  Cin,  H, W, Cout, Cout, 1);
    conv3x3(s, tA, ps[1], nullptr, tB,  B, Cout, Cout, H, W, Cout, Cout, 1);
    conv3x3(s, tB, ps[2], nullptr, dst, B, Cout, Cout, H, W, Cout, Cout, 1);
}

extern "C" void kernel_launch(void* const* d_in, const int* in_sizes, int n_in,
                              void* d_out, int out_size, void* d_ws, size_t ws_size,
                              hipStream_t stream)
{
    (void)in_sizes; (void)n_in; (void)out_size; (void)ws_size;

    const int B = 2, Hin = 334, Win = 334;
    const int HP = 384;                 // padded spatial (ceil128 of 334+50)
    const int HR = 434;                 // replication-padded for sepconv
    const int PAD = 25;

    const float* t1 = (const float*)d_in[0];
    const float* t2 = (const float*)d_in[1];

    int li = 2;
    auto grab = [&](ConvP* dst, int n) {
        for (int k = 0; k < n; ++k) {
            dst[k].w = (const float*)d_in[li];
            dst[k].b = (const float*)d_in[li + 1];
            li += 2;
        }
    };
    ConvP c1p[3], c2p[3], c3p[3], c4p[3], c5p[3];
    ConvP d5p[3], u5p[1], d4p[3], u4p[1], d3p[3], u3p[1], d2p[3], u2p[1];
    ConvP v1p[4], v2p[4], h1p[4], h2p[4];
    grab(c1p, 3); grab(c2p, 3); grab(c3p, 3); grab(c4p, 3); grab(c5p, 3);
    grab(d5p, 3); grab(u5p, 1); grab(d4p, 3); grab(u4p, 1);
    grab(d3p, 3); grab(u3p, 1); grab(d2p, 3); grab(u2p, 1);
    grab(v1p, 4); grab(v2p, 4); grab(h1p, 4); grab(h2p, 4);

    // ---- workspace bump allocator -----------------------------------------
    char* wbase = (char*)d_ws;
    size_t off = 0;
    auto allocB = [&](size_t bytes) -> void* {
        void* p = wbase + off;
        off += (bytes + 255) & ~(size_t)255;
        return p;
    };
    auto allocH = [&](long long elems) -> half_t* {
        return (half_t*)allocB((size_t)elems * sizeof(half_t));
    };
    auto el = [&](int c, int h) -> long long { return (long long)B * c * h * h; };

    half_t* join = allocH(el(16, 384));
    float*  rep1 = (float*)allocB((size_t)B * 3 * HR * HR * sizeof(float));
    float*  rep2 = (float*)allocB((size_t)B * 3 * HR * HR * sizeof(float));

    half_t* A1a = allocH(el(32, 384)); half_t* A1b = allocH(el(32, 384));
    half_t* p1  = allocH(el(32, 192));
    half_t* A2a = allocH(el(64, 192)); half_t* A2b = allocH(el(64, 192));
    half_t* c2  = allocH(el(64, 192)); half_t* p2  = allocH(el(64, 96));
    half_t* A3a = allocH(el(128, 96)); half_t* A3b = allocH(el(128, 96));
    half_t* c3  = allocH(el(128, 96)); half_t* p3  = allocH(el(128, 48));
    half_t* A4a = allocH(el(256, 48)); half_t* A4b = allocH(el(256, 48));
    half_t* c4  = allocH(el(256, 48)); half_t* p4  = allocH(el(256, 24));
    half_t* A5a = allocH(el(512, 24)); half_t* A5b = allocH(el(512, 24));
    half_t* c5  = allocH(el(512, 24)); half_t* p5  = allocH(el(512, 12));
    half_t* D5a = allocH(el(512, 12)); half_t* D5b = allocH(el(512, 12));
    half_t* U5  = allocH(el(512, 24)); half_t* cb5 = allocH(el(512, 24));
    half_t* D4a = allocH(el(256, 24)); half_t* D4b = allocH(el(256, 24));
    half_t* U4  = allocH(el(256, 48)); half_t* cb4 = allocH(el(256, 48));
    half_t* D3a = allocH(el(128, 48)); half_t* D3b = allocH(el(128, 48));
    half_t* U3  = allocH(el(128, 96)); half_t* cb3 = allocH(el(128, 96));
    half_t* D2a = allocH(el(64, 96));  half_t* D2b = allocH(el(64, 96));
    half_t* U2  = allocH(el(64, 192)); half_t* comb = allocH(el(64, 192));
    half_t* S1  = allocH(el(64, 192)); half_t* S2  = allocH(el(64, 192));
    half_t* S51 = allocH(el(64, 192));   // 51 real + 13 zero-pad channels
    half_t* Sup = allocH(el(64, 384));   // upsampled, padded channels stay 0
    half_t* cV  = allocH(el(51, 384)); half_t* cH = allocH(el(51, 384));

    float* outp = (float*)d_out;

    // ---- pads + channel-pad zero fill --------------------------------------
    pad_join_kernel<<<grid1d(el(16, 384)), 256, 0, stream>>>(t1, t2, join, B, Hin, Win, HP, HP, PAD);
    long long tr = (long long)B * 3 * HR * HR;
    pad_rep_kernel<<<grid1d(tr), 256, 0, stream>>>(t1, rep1, B, Hin, Win, HR, HR, 2 * PAD);
    pad_rep_kernel<<<grid1d(tr), 256, 0, stream>>>(t2, rep2, B, Hin, Win, HR, HR, 2 * PAD);
    fill0_kernel<<<grid1d(el(64, 192)), 256, 0, stream>>>(S51, el(64, 192));

    // ---- encoder -----------------------------------------------------------
    conv3x3(stream, join, c1p[0], nullptr, A1a, B, 16, 6, 384, 384, 32, 32, 1);
    conv3x3(stream, A1a, c1p[1], nullptr, A1b, B, 32, 32, 384, 384, 32, 32, 1);
    conv3x3(stream, A1b, c1p[2], nullptr, A1a, B, 32, 32, 384, 384, 32, 32, 1);
    avgpool2_kernel<<<grid1d(el(32, 192)), 256, 0, stream>>>(A1a, p1, B, 32, 384, 384);
    basic3(stream, p1, c2p, A2a, A2b, c2, B, 32, 192, 192, 64);
    avgpool2_kernel<<<grid1d(el(64, 96)), 256, 0, stream>>>(c2, p2, B, 64, 192, 192);
    basic3(stream, p2, c3p, A3a, A3b, c3, B, 64, 96, 96, 128);
    avgpool2_kernel<<<grid1d(el(128, 48)), 256, 0, stream>>>(c3, p3, B, 128, 96, 96);
    basic3(stream, p3, c4p, A4a, A4b, c4, B, 128, 48, 48, 256);
    avgpool2_kernel<<<grid1d(el(256, 24)), 256, 0, stream>>>(c4, p4, B, 256, 48, 48);
    basic3(stream, p4, c5p, A5a, A5b, c5, B, 256, 24, 24, 512);
    avgpool2_kernel<<<grid1d(el(512, 12)), 256, 0, stream>>>(c5, p5, B, 512, 24, 24);

    // ---- decoder (skip-add fused into conv epilogue) -----------------------
    basic3(stream, p5, d5p, D5a, D5b, D5a, B, 512, 12, 12, 512);
    upsample2_ac_kernel<<<grid1d(el(512, 24)), 256, 0, stream>>>(D5a, U5, B, 512, 12, 12);
    conv3x3(stream, U5, u5p[0], c5, cb5, B, 512, 512, 24, 24, 512, 512, 1);

    basic3(stream, cb5, d4p, D4a, D4b, D4a, B, 512, 24, 24, 256);
    upsample2_ac_kernel<<<grid1d(el(256, 48)), 256, 0, stream>>>(D4a, U4, B, 256, 24, 24);
    conv3x3(stream, U4, u4p[0], c4, cb4, B, 256, 256, 48, 48, 256, 256, 1);

    basic3(stream, cb4, d3p, D3a, D3b, D3a, B, 256, 48, 48, 128);
    upsample2_ac_kernel<<<grid1d(el(128, 96)), 256, 0, stream>>>(D3a, U3, B, 128, 48, 48);
    conv3x3(stream, U3, u3p[0], c3, cb3, B, 128, 128, 96, 96, 128, 128, 1);

    basic3(stream, cb3, d2p, D2a, D2b, D2a, B, 128, 96, 96, 64);
    upsample2_ac_kernel<<<grid1d(el(64, 192)), 256, 0, stream>>>(D2a, U2, B, 64, 96, 96);
    conv3x3(stream, U2, u2p[0], c2, comb, B, 64, 64, 192, 192, 64, 64, 1);

    // ---- subnets + separable application ----------------------------------
    auto subnet = [&](const ConvP* ps, half_t* coeff) {
        conv3x3(stream, comb, ps[0], nullptr, S1, B, 64, 64, 192, 192, 64, 64, 1);
        conv3x3(stream, S1, ps[1], nullptr, S2, B, 64, 64, 192, 192, 64, 64, 1);
        conv3x3(stream, S2, ps[2], nullptr, S51, B, 64, 64, 192, 192, 51, 64, 1);
        upsample2_ac_kernel<<<grid1d(el(64, 384)), 256, 0, stream>>>(S51, Sup, B, 64, 192, 192);
        conv3x3(stream, Sup, ps[3], nullptr, coeff, B, 64, 51, 384, 384, 51, 51, 0);
    };

    dim3 sgrid((Win + SEP_TS - 1) / SEP_TS, (Hin + SEP_TS - 1) / SEP_TS, B * 3);
    dim3 sblk(SEP_TS, SEP_TS);

    subnet(v1p, cV);
    subnet(h1p, cH);
    sepconv_kernel<<<sgrid, sblk, 0, stream>>>(rep1, cV, cH, outp,
                                               B, Hin, Win, HR, HR, HP, 0);
    subnet(v2p, cV);
    subnet(h2p, cH);
    sepconv_kernel<<<sgrid, sblk, 0, stream>>>(rep2, cV, cH, outp,
                                               B, Hin, Win, HR, HR, HP, 1);
}
